// GenreAgnosticInference_58428735095185
// MI455X (gfx1250) — matrix operand order repaired and checked by reading, not compiled, not run on hardware
//
#include <hip/hip_runtime.h>
#include <hip/hip_bf16.h>

typedef __attribute__((ext_vector_type(16))) __bf16 v16bf;
typedef __attribute__((ext_vector_type(8)))  __bf16 v8bf;
typedef __attribute__((ext_vector_type(8)))  float  v8f;
typedef __attribute__((ext_vector_type(4)))  int    v4i;

#define BATCH 128
#define EDIM  300
#define HID   300
#define PLEN  32
#define HLEN  32
#define EP    320          // E / HID padded to multiple of 32
#define KCAT  640          // concat K: [left 320 | lower 320]
#define NG    2400         // 8*HID gate width (logical)
#define NGR   2432         // NG padded to multiple of 128 (N-tile) for guard-free staging
#define EPR   384          // EP padded to multiple of 128
#define NTILE 128          // N tile per workgroup

// ---- CDNA5 async global->LDS staging (guarded: falls back to sync path) ----
#if __has_builtin(__builtin_amdgcn_global_load_async_to_lds_b128)
#define HAVE_ASYNC_LDS 1
#define ASYNC_B128(g, l, off)                                                  \
    __builtin_amdgcn_global_load_async_to_lds_b128(                            \
        (__attribute__((address_space(1))) v4i*)(g),                           \
        (__attribute__((address_space(3))) v4i*)(l), (off), 0)
#if __has_builtin(__builtin_amdgcn_s_wait_asynccnt)
#define WAIT_ASYNC(n) __builtin_amdgcn_s_wait_asynccnt(n)
#else
#define WAIT_ASYNC(n) asm volatile("s_wait_asynccnt %0" ::"i"(n) : "memory")
#endif
#else
#define HAVE_ASYNC_LDS 0
#endif

// ---------------------------------------------------------------------------
// Generic WMMA GEMM:  C[z] = op( A[z][128,K] @ W[N,K]^T + bias + add0[z] + add1[z] )
// A may be stitched from two sources at k = ksplit (for hcat = [left|lower]).
// W buffers are row-padded so staging never needs a bounds check.
// 256 threads = 8 waves (2x4), wave tile 64x32, per-wave 4x2 wmma accumulators.
// ---------------------------------------------------------------------------
struct GemmArgs {
    const __bf16* A0; long long a0zs;
    const __bf16* A1; long long a1zs;       // used for k >= ksplit
    int lda; int K; int ksplit;             // K, ksplit multiples of 32
    const __bf16* W; int N;                 // W is [>=ceil128(N)][K] bf16, K-padded
    const float* bias0; const float* bias1; int nbias;
    const float* add0; long long add0zs;    // optional f32 C-init, [128][addld]
    const float* add1; long long add1zs;
    int addld;
    void* C; long long czs; int ldc;        // czs/ldc in elements of out type
    int cbf16; int relu;
};

__launch_bounds__(256)
__global__ void k_wmma_gemm(GemmArgs ga) {
    __shared__ __attribute__((aligned(32))) __bf16 lA[2][128][40];
    __shared__ __attribute__((aligned(32))) __bf16 lB[2][128][40];

    const int z  = blockIdx.z;
    const int n0 = blockIdx.x * NTILE;

    const __bf16* A0 = ga.A0 + (long long)z * ga.a0zs;
    const __bf16* A1 = ga.A1 + (long long)z * ga.a1zs;
    const float* add0 = ga.add0 ? ga.add0 + (long long)z * ga.add0zs : nullptr;
    const float* add1 = ga.add1 ? ga.add1 + (long long)z * ga.add1zs : nullptr;

    const int t    = threadIdx.x;
    const int lane = t & 31;
    const int wid  = t >> 5;
    const int wm   = (wid >> 2) * 64;   // wave row offset (0 / 64)
    const int wn   = (wid & 3) * 32;    // wave col offset (0/32/64/96)
    const int half = lane >> 4;
    const int l16  = lane & 15;
    const int srow = t >> 1;            // staging: row 0..127
    const int skc  = (t & 1) * 16;      // staging: k-chunk 0 / 16

    v8f acc[4][2];
#pragma unroll
    for (int a = 0; a < 4; ++a)
#pragma unroll
        for (int b = 0; b < 2; ++b)
#pragma unroll
            for (int e = 0; e < 8; ++e) acc[a][b][e] = 0.f;

    // per-thread staging source base addresses
    const __bf16* wsrc = ga.W + (long long)(n0 + srow) * ga.K + skc;

    auto asrc_for = [&](int k0) -> const __bf16* {
        return (k0 < ga.ksplit)
            ? (A0 + (long long)srow * ga.lda + k0 + skc)
            : (A1 + (long long)srow * ga.lda + (k0 - ga.ksplit) + skc);
    };

    auto compute = [&](int buf) {
        // B fragments: lane l16 = column n, half selects K 0-15 / 16-31
        v16bf bfrag[2];
#pragma unroll
        for (int nn = 0; nn < 2; ++nn) {
            const int row = wn + nn * 16 + l16;
            bfrag[nn] = *(const v16bf*)&lB[buf][row][half * 16];
        }
        // A fragments: lane l16 = row m; half selects K 0-15 / 16-31
        v16bf afrag[4];
#pragma unroll
        for (int mm = 0; mm < 4; ++mm) {
            const int row = wm + mm * 16 + l16;
            v8bf c0 = *(const v8bf*)&lA[buf][row][half * 16];
            v8bf c1 = *(const v8bf*)&lA[buf][row][half * 16 + 8];
            v16bf a;
#pragma unroll
            for (int e = 0; e < 8; ++e) { a[e] = c0[e]; a[8 + e] = c1[e]; }
            afrag[mm] = a;
        }
#pragma unroll
        for (int mm = 0; mm < 4; ++mm)
#pragma unroll
            for (int nn = 0; nn < 2; ++nn)
                acc[mm][nn] = __builtin_amdgcn_wmma_f32_16x16x32_bf16(
                    false, afrag[mm], false, bfrag[nn],
                    (short)0, acc[mm][nn], false, false);
    };

    const int KT = ga.K / 32;

#if HAVE_ASYNC_LDS
    // -------- async double-buffered pipeline: copy k+1 overlaps wmma on k ----
    {
        const __bf16* as = asrc_for(0);
        ASYNC_B128(as, &lA[0][srow][skc], 0);
        ASYNC_B128(as, &lA[0][srow][skc], 16);
        ASYNC_B128(wsrc, &lB[0][srow][skc], 0);
        ASYNC_B128(wsrc, &lB[0][srow][skc], 16);
    }
    for (int kt = 0; kt < KT; ++kt) {
        const int buf = kt & 1;
        if (kt + 1 < KT) {
            const int k0n = (kt + 1) * 32;
            const __bf16* as = asrc_for(k0n);
            const __bf16* bs = wsrc + k0n;
            ASYNC_B128(as, &lA[buf ^ 1][srow][skc], 0);
            ASYNC_B128(as, &lA[buf ^ 1][srow][skc], 16);
            ASYNC_B128(bs, &lB[buf ^ 1][srow][skc], 0);
            ASYNC_B128(bs, &lB[buf ^ 1][srow][skc], 16);
            WAIT_ASYNC(4);          // tile kt's 4 copies done (per thread)
        } else {
            WAIT_ASYNC(0);
        }
        __syncthreads();            // all threads' copies for tile kt visible
        compute(buf);
        __syncthreads();            // compute done before buf is overwritten
    }
#else
    // -------- synchronous fallback ----
    for (int kt = 0; kt < KT; ++kt) {
        const int k0 = kt * 32;
        *(v16bf*)&lA[0][srow][skc] = *(const v16bf*)asrc_for(k0);
        *(v16bf*)&lB[0][srow][skc] = *(const v16bf*)(wsrc + k0);
        __syncthreads();
        compute(0);
        __syncthreads();
    }
#endif

    // ---- epilogue: C/D layout (VGPR v: lanes0-15 M=v, lanes16-31 M=v+8; N=l16) ----
#pragma unroll
    for (int mm = 0; mm < 4; ++mm) {
#pragma unroll
        for (int nn = 0; nn < 2; ++nn) {
            const int n = n0 + wn + nn * 16 + l16;
            if (n >= ga.N) continue;
            float bv = 0.f;
            if (n < ga.nbias) {
                bv = ga.bias0[n];
                if (ga.bias1) bv += ga.bias1[n];
            }
#pragma unroll
            for (int vi = 0; vi < 8; ++vi) {
                const int m = wm + mm * 16 + vi + half * 8;
                float val = acc[mm][nn][vi] + bv;
                if (add0) val += add0[(long long)m * ga.addld + n];
                if (add1) val += add1[(long long)m * ga.addld + n];
                if (ga.relu) val = fmaxf(val, 0.f);
                if (ga.cbf16) {
                    __bf16* C = (__bf16*)ga.C + (long long)z * ga.czs;
                    C[(long long)m * ga.ldc + n] = (__bf16)val;
                } else {
                    float* C = (float*)ga.C + (long long)z * ga.czs;
                    C[(long long)m * ga.ldc + n] = val;
                }
            }
        }
    }
}

// ---------------------------------------------------------------------------
// Embedding gather -> bf16, [seq][128][EP] with zero K-pad
// ---------------------------------------------------------------------------
__global__ void k_embed(const int* __restrict__ tok, const float* __restrict__ emb,
                        __bf16* __restrict__ out, int L) {
    long long idx = (long long)blockIdx.x * blockDim.x + threadIdx.x;
    long long tot = (long long)L * BATCH * EP;
    if (idx >= tot) return;
    int k = (int)(idx % EP);
    long long r = idx / EP;
    int b = (int)(r % BATCH);
    int s = (int)(r / BATCH);
    float v = 0.f;
    if (k < EDIM) v = emb[(long long)tok[b * L + s] * EDIM + k];
    out[idx] = (__bf16)v;
}

// ---------------------------------------------------------------------------
// Weight repack f32 -> padded bf16 (rows >= Nin and pad cols are zero).
// split=0: out[n][k] = (n<Nin && k<take) ? in[n*Kin + colOff + k] : 0
// split=1: Kin=600 -> KP=640, halves land at 0..299 and 320..619
// ---------------------------------------------------------------------------
__global__ void k_pack(const float* __restrict__ in, __bf16* __restrict__ out,
                       int Nin, int Kin, int NP, int KP, int colOff, int take, int split) {
    long long idx = (long long)blockIdx.x * blockDim.x + threadIdx.x;
    long long tot = (long long)NP * KP;
    if (idx >= tot) return;
    int n = (int)(idx / KP);
    int k = (int)(idx % KP);
    float v = 0.f;
    if (n < Nin) {
        if (split) {
            if (k < 300)                  v = in[(long long)n * Kin + k];
            else if (k >= 320 && k < 620) v = in[(long long)n * Kin + (k - 20)];
        } else if (k < take) {
            v = in[(long long)n * Kin + colOff + k];
        }
    }
    out[idx] = (__bf16)v;
}

__global__ void k_zero_bf16(__bf16* __restrict__ p, long long n) {
    long long idx = (long long)blockIdx.x * blockDim.x + threadIdx.x;
    if (idx < n) p[idx] = (__bf16)0.f;
}

// ---------------------------------------------------------------------------
// LSTM cell elementwise: gates[z][128][2400] -> h_full/c_full (bf16, K-padded 640)
// ccat element n<300 comes from left_c (slot j+1), else lower_c (slot j).
// ---------------------------------------------------------------------------
__global__ void k_cellact(const float* __restrict__ gates,
                          const __bf16* __restrict__ Sc,
                          __bf16* __restrict__ hf, __bf16* __restrict__ cf,
                          int d, int i0) {
    const int z = blockIdx.z;
    const int j = d - i0 - z;
    long long idx = (long long)blockIdx.x * blockDim.x + threadIdx.x;
    if (idx >= (long long)BATCH * 600) return;
    const int b = (int)(idx / 600);
    const int n = (int)(idx % 600);
    const float* g = gates + ((long long)z * BATCH + b) * NG;
    const float ig = g[n];
    const float fg = g[600 + n];
    const float gg = g[1200 + n];
    const float og = g[1800 + n];
    const float ccat = (n < 300)
        ? (float)Sc[((long long)(j + 1) * BATCH + b) * EP + n]
        : (float)Sc[((long long)j * BATCH + b) * EP + (n - 300)];
    const float si = 1.f / (1.f + __expf(-ig));
    const float sf = 1.f / (1.f + __expf(-fg));
    const float so = 1.f / (1.f + __expf(-og));
    const float c = sf * ccat + si * tanhf(gg);
    const float h = so * tanhf(c);
    const int kc = (n < 300) ? n : n + 20;      // matches split weight packing
    const long long o = ((long long)z * BATCH + b) * KCAT + kc;
    hf[o] = (__bf16)h;
    cf[o] = (__bf16)c;
}

// ---------------------------------------------------------------------------
// Head: logits = t2 @ W3^T + b3 ; softmax over 3 classes
// ---------------------------------------------------------------------------
__global__ void k_head(const __bf16* __restrict__ t2, const float* __restrict__ W3,
                       const float* __restrict__ b3, float* __restrict__ out) {
    const int b = threadIdx.x;
    if (b >= BATCH) return;
    float l[3];
    for (int c = 0; c < 3; ++c) {
        float s = b3[c];
        for (int k = 0; k < 300; ++k)
            s += (float)t2[(long long)b * EP + k] * W3[c * 300 + k];
        l[c] = s;
    }
    const float mx = fmaxf(l[0], fmaxf(l[1], l[2]));
    const float e0 = __expf(l[0] - mx), e1 = __expf(l[1] - mx), e2 = __expf(l[2] - mx);
    const float inv = 1.f / (e0 + e1 + e2);
    out[b * 3 + 0] = e0 * inv;
    out[b * 3 + 1] = e1 * inv;
    out[b * 3 + 2] = e2 * inv;
}

// ---------------------------------------------------------------------------
extern "C" void kernel_launch(void* const* d_in, const int* in_sizes, int n_in,
                              void* d_out, int out_size, void* d_ws, size_t ws_size,
                              hipStream_t stream) {
    (void)in_sizes; (void)n_in; (void)out_size; (void)ws_size;
    const int*   premise = (const int*)d_in[0];
    const int*   hypoth  = (const int*)d_in[1];
    const float* emb     = (const float*)d_in[2];
    const float* W_ih    = (const float*)d_in[3];
    const float* W_hh    = (const float*)d_in[4];
    const float* b_ih    = (const float*)d_in[5];
    const float* b_hh    = (const float*)d_in[6];
    const float* W_ch    = (const float*)d_in[7];
    const float* b_ch    = (const float*)d_in[8];
    const float* W_cc    = (const float*)d_in[9];
    const float* b_cc    = (const float*)d_in[10];
    const float* W1      = (const float*)d_in[11];
    const float* b1      = (const float*)d_in[12];
    const float* W2      = (const float*)d_in[13];
    const float* b2      = (const float*)d_in[14];
    const float* W3      = (const float*)d_in[15];
    const float* b3      = (const float*)d_in[16];
    float* out = (float*)d_out;

    // ---- carve workspace (all 256B aligned) ----
    char* ws = (char*)d_ws;
    size_t off = 0;
    auto carve = [&](size_t bytes) -> char* {
        char* p = ws + off;
        off += (bytes + 255) & ~(size_t)255;
        return p;
    };
    __bf16* peb  = (__bf16*)carve((size_t)PLEN * BATCH * EP * 2);
    __bf16* heb  = (__bf16*)carve((size_t)HLEN * BATCH * EP * 2);
    __bf16* Wp   = (__bf16*)carve((size_t)NGR * EP * 2);
    __bf16* Whyp = (__bf16*)carve((size_t)NGR * EP * 2);
    __bf16* Whh  = (__bf16*)carve((size_t)NGR * KCAT * 2);
    __bf16* Wchp = (__bf16*)carve((size_t)EPR * KCAT * 2);
    __bf16* Wccp = (__bf16*)carve((size_t)EPR * KCAT * 2);
    __bf16* W1p  = (__bf16*)carve((size_t)EPR * EP * 2);
    __bf16* W2p  = (__bf16*)carve((size_t)EPR * EP * 2);
    float*  Xp   = (float*)carve((size_t)PLEN * BATCH * NG * 4);
    float*  Xh   = (float*)carve((size_t)HLEN * BATCH * NG * 4);
    float*  gbuf = (float*)carve((size_t)32 * BATCH * NG * 4);
    __bf16* hf   = (__bf16*)carve((size_t)32 * BATCH * KCAT * 2);
    __bf16* cf   = (__bf16*)carve((size_t)32 * BATCH * KCAT * 2);
    __bf16* Sh   = (__bf16*)carve((size_t)33 * BATCH * EP * 2);
    __bf16* Sc   = (__bf16*)carve((size_t)33 * BATCH * EP * 2);
    __bf16* t1   = (__bf16*)carve((size_t)BATCH * EP * 2);
    __bf16* t2   = (__bf16*)carve((size_t)BATCH * EP * 2);

    auto zero = [&](__bf16* p, long long n) {
        k_zero_bf16<<<dim3((unsigned)((n + 255) / 256)), 256, 0, stream>>>(p, n);
    };
    // states must start at zero every call (boundary rows/cols + pad columns)
    zero(Sh, (long long)33 * BATCH * EP);
    zero(Sc, (long long)33 * BATCH * EP);
    zero(hf, (long long)32 * BATCH * KCAT);
    zero(cf, (long long)32 * BATCH * KCAT);

    // ---- embedding gather + weight repack ----
    {
        long long tot = (long long)PLEN * BATCH * EP;
        k_embed<<<dim3((unsigned)((tot + 255) / 256)), 256, 0, stream>>>(premise, emb, peb, PLEN);
        k_embed<<<dim3((unsigned)((tot + 255) / 256)), 256, 0, stream>>>(hypoth, emb, heb, HLEN);
    }
    auto pack = [&](const float* in, __bf16* o, int Nin, int Kin, int NP, int KP,
                    int colOff, int take, int split) {
        long long tot = (long long)NP * KP;
        k_pack<<<dim3((unsigned)((tot + 255) / 256)), 256, 0, stream>>>(
            in, o, Nin, Kin, NP, KP, colOff, take, split);
    };
    pack(W_ih, Wp,   NG,  600, NGR, EP,   0,   300, 0);   // premise half of W_ih
    pack(W_ih, Whyp, NG,  600, NGR, EP,   300, 300, 0);   // hypothesis half
    pack(W_hh, Whh,  NG,  600, NGR, KCAT, 0,   0,   1);   // split-concat layout
    pack(W_ch, Wchp, HID, 600, EPR, KCAT, 0,   0,   1);
    pack(W_cc, Wccp, HID, 600, EPR, KCAT, 0,   0,   1);
    pack(W1,   W1p,  HID, 300, EPR, EP,   0,   300, 0);
    pack(W2,   W2p,  HID, 300, EPR, EP,   0,   300, 0);

    auto gemm = [&](GemmArgs ga, int ntiles, int nz) {
        k_wmma_gemm<<<dim3(ntiles, 1, nz), 256, 0, stream>>>(ga);
    };
    const long long SLOT = (long long)BATCH * EP;    // one state slot, elements
    const long long GROW = (long long)BATCH * NG;

    // ---- hoisted input projections: Xp[i] = pe_i @ Wp^T, Xh[j] = he_j @ Whyp^T ----
    {
        GemmArgs ga = {};
        ga.A0 = peb; ga.a0zs = SLOT; ga.A1 = peb; ga.a1zs = SLOT;
        ga.lda = EP; ga.K = EP; ga.ksplit = EP;
        ga.W = Wp; ga.N = NG;
        ga.bias0 = nullptr; ga.bias1 = nullptr; ga.nbias = 0;
        ga.add0 = nullptr; ga.add1 = nullptr; ga.addld = NG;
        ga.C = Xp; ga.czs = GROW; ga.ldc = NG; ga.cbf16 = 0; ga.relu = 0;
        gemm(ga, NGR / NTILE, PLEN);
        ga.A0 = heb; ga.A1 = heb; ga.W = Whyp; ga.C = Xh;
        gemm(ga, NGR / NTILE, HLEN);
    }

    // ---- 63-step anti-diagonal wavefront over the 32x32 grid ----
    for (int d = 0; d < PLEN + HLEN - 1; ++d) {
        const int i0 = (d < HLEN) ? 0 : d - (HLEN - 1);
        const int i1 = (d < PLEN) ? d : PLEN - 1;
        const int nc = i1 - i0 + 1;
        const int j0 = d - i0;   // z=0 cell's column; j decreases with z

        // gates[z] = Xp[i0+z] + Xh[j0-z] + (b_ih+b_hh) + [Sh(j+1)|Sh(j)] @ Whh^T
        {
            GemmArgs ga = {};
            ga.A0 = Sh + (long long)(j0 + 1) * SLOT; ga.a0zs = -SLOT;   // left_h
            ga.A1 = Sh + (long long)j0 * SLOT;       ga.a1zs = -SLOT;   // lower_h
            ga.lda = EP; ga.K = KCAT; ga.ksplit = EP;
            ga.W = Whh; ga.N = NG;
            ga.bias0 = b_ih; ga.bias1 = b_hh; ga.nbias = NG;
            ga.add0 = Xp + (long long)i0 * GROW; ga.add0zs = GROW;
            ga.add1 = Xh + (long long)j0 * GROW; ga.add1zs = -GROW;
            ga.addld = NG;
            ga.C = gbuf; ga.czs = GROW; ga.ldc = NG; ga.cbf16 = 0; ga.relu = 0;
            gemm(ga, NGR / NTILE, nc);
        }
        // elementwise LSTM cell -> h_full/c_full (bf16, padded K=640)
        {
            long long tot = (long long)BATCH * 600;
            k_cellact<<<dim3((unsigned)((tot + 255) / 256), 1, nc), 256, 0, stream>>>(
                gbuf, Sc, hf, cf, d, i0);
        }
        // condense: Sh[j+1] = hf @ Wch^T + b_ch ; Sc[j+1] = cf @ Wcc^T + b_cc
        {
            GemmArgs ga = {};
            ga.A0 = hf; ga.a0zs = (long long)BATCH * KCAT;
            ga.A1 = hf; ga.a1zs = (long long)BATCH * KCAT;
            ga.lda = KCAT; ga.K = KCAT; ga.ksplit = KCAT;
            ga.W = Wchp; ga.N = EP;
            ga.bias0 = b_ch; ga.bias1 = nullptr; ga.nbias = HID;
            ga.add0 = nullptr; ga.add1 = nullptr; ga.addld = EP;
            ga.C = Sh + (long long)(j0 + 1) * SLOT; ga.czs = -SLOT;
            ga.ldc = EP; ga.cbf16 = 1; ga.relu = 0;
            gemm(ga, EPR / NTILE, nc);
            ga.A0 = cf; ga.A1 = cf; ga.W = Wccp;
            ga.bias0 = b_cc;
            ga.C = Sc + (long long)(j0 + 1) * SLOT;
            gemm(ga, EPR / NTILE, nc);
        }
    }

    // ---- MLP head: t1 = relu(h @ W1^T + b1); t2 = relu(t1 @ W2^T + b2) ----
    {
        GemmArgs ga = {};
        ga.A0 = Sh + (long long)HLEN * SLOT; ga.a0zs = 0;   // state (P-1, Hn-1)
        ga.A1 = ga.A0; ga.a1zs = 0;
        ga.lda = EP; ga.K = EP; ga.ksplit = EP;
        ga.W = W1p; ga.N = EP;
        ga.bias0 = b1; ga.bias1 = nullptr; ga.nbias = HID;
        ga.add0 = nullptr; ga.add1 = nullptr; ga.addld = EP;
        ga.C = t1; ga.czs = 0; ga.ldc = EP; ga.cbf16 = 1; ga.relu = 1;
        gemm(ga, EPR / NTILE, 1);
        ga.A0 = t1; ga.A1 = t1; ga.W = W2p; ga.bias0 = b2; ga.C = t2;
        gemm(ga, EPR / NTILE, 1);
    }
    k_head<<<1, 128, 0, stream>>>(t2, W3, b3, out);
}
